// GraphNN_171798692410
// MI455X (gfx1250) — matrix operand order repaired and checked by reading, not compiled, not run on hardware
//
#include <hip/hip_runtime.h>

// Problem constants (match reference)
#define B_ 32
#define N_ 1000
#define E_ 2000
#define H_ 256
#define HOPS_ 3

typedef __attribute__((ext_vector_type(16))) __bf16 v16bf;
typedef __attribute__((ext_vector_type(8)))  float  v8f;

__device__ __forceinline__ unsigned short f2bf(float f) {
  union { float f; unsigned u; } x; x.f = f;
  unsigned r = x.u + 0x7FFFu + ((x.u >> 16) & 1u);   // round-to-nearest-even
  return (unsigned short)(r >> 16);
}
__device__ __forceinline__ unsigned pack_bf16(float lo, float hi) {
  return (unsigned)f2bf(lo) | ((unsigned)f2bf(hi) << 16);
}
__device__ __forceinline__ float sigm(float x) { return 1.f / (1.f + __expf(-x)); }

enum { AS_F32 = 0, AS_BF16 = 1 };
enum { AM_SINGLE = 0, AM_MUL = 1, AM_SUB = 2 };
enum { EPI_NONE = 0, EPI_BIAS = 1, EPI_BIAS_RELU = 2 };

// ---------------------------------------------------------------------------
// Batched GEMM: C[b][M,256] (+)= f(A,A2)[b][M,K] * Bm[b][K,256]
//   ASRC   : A stored as f32 or pre-converted bf16 (adjacency)
//   TRANS_A: A element (m,k) read as A[k*lda + m]
//   AMODE  : A_eff = A, A*A2, A-A2 (gated-fusion concat terms; f32 only)
//   ACCUM  : C += result ; EPI: bias / bias+relu epilogue
// Block tile 64x256 (grid.y==1 -> A slab read exactly once), 256 threads =
// 8 wave32 waves, each wave 32x64 = 2x4 v_wmma_f32_16x16x32_bf16, K tile 32.
// Boundary staging is branchless: clamped addresses + select-to-zero, so no
// exec-mask divergence anywhere in the staging loops.
// ---------------------------------------------------------------------------
template<int ASRC, bool TRANS_A, int AMODE, bool ACCUM, int EPI>
__global__ __launch_bounds__(256) void gemm_bf16_wmma(
    const void* __restrict__ Ap, const float* __restrict__ A2, int lda, size_t sA,
    const float* __restrict__ Bm, int ldb, size_t sB,
    float* __restrict__ C, int ldc, size_t sC,
    const float* __restrict__ bias,
    int M, int Ncols, int K)
{
  constexpr int BM = 64, BN = 256, BK = 32;
  __shared__ unsigned AsU[BM * (BK / 2)];   // [m][kpair], 16 u32 per row
  __shared__ unsigned BsU[BN * (BK / 2)];   // [n][kpair], 16 u32 per row

  const int b = blockIdx.z;
  const unsigned short* Ab = (const unsigned short*)Ap + (ASRC == AS_BF16 ? (size_t)b * sA : 0);
  const float*          Af = (const float*)Ap          + (ASRC == AS_F32  ? (size_t)b * sA : 0);
  if (AMODE != AM_SINGLE) A2 += (size_t)b * sA;
  Bm += (size_t)b * sB;
  C  += (size_t)b * sC;

  const int m0 = blockIdx.x * BM, n0 = blockIdx.y * BN;
  const int tid  = threadIdx.x;
  const int lane = tid & 31, wid = tid >> 5;
  const int wm = (wid & 1) * 32, wn = (wid >> 1) * 64;   // wave's 32x64 sub-tile
  const int half = lane >> 4, l16 = lane & 15;

  v8f acc[2][4] = {};

  for (int k0 = 0; k0 < K; k0 += BK) {
    const bool interior = (m0 + BM <= M) && (k0 + BK <= K) && (n0 + BN <= Ncols);

    // ---------------- stage A tile ----------------
    if (ASRC == AS_BF16 && !TRANS_A && interior) {
      // fast path: straight uint4 copy (global_load_b128 -> ds_store_b128)
      #pragma unroll
      for (int i = 0; i < 1; ++i) {  // BM*4 == 256 == blockDim
        int r = tid >> 2, kq = (tid & 3) * 4;
        const uint4* src = (const uint4*)(Ab + (size_t)(m0 + r) * lda + k0) + (kq >> 2);
        *((uint4*)&AsU[r * 16 + kq]) = *src;
        // prefetch next K tile of the streamed A slab (global_prefetch_b8)
        if (k0 + BK < K)
          __builtin_prefetch((const char*)(Ab + (size_t)(m0 + r) * lda + k0 + BK), 0, 0);
      }
    } else {
      for (int i = tid; i < BM * 16; i += 256) {
        int r = i >> 4, kp = i & 15;
        int gm = m0 + r, gk = k0 + 2 * kp;
        // branchless guards: clamp indices, load unconditionally, select to 0
        const bool vm = gm < M;
        const bool v0 = vm & (gk < K), v1 = vm & ((gk + 1) < K);
        const int cm  = vm ? gm : M - 1;
        const int ck0 = (gk     < K) ? gk     : K - 1;
        const int ck1 = (gk + 1 < K) ? gk + 1 : K - 1;
        unsigned v;
        if (ASRC == AS_BF16) {
          unsigned short lo = Ab[TRANS_A ? (size_t)ck0 * lda + cm : (size_t)cm * lda + ck0];
          unsigned short hi = Ab[TRANS_A ? (size_t)ck1 * lda + cm : (size_t)cm * lda + ck1];
          v = (v0 ? (unsigned)lo : 0u) | (v1 ? ((unsigned)hi << 16) : 0u);
        } else {
          size_t i0 = TRANS_A ? (size_t)ck0 * lda + cm : (size_t)cm * lda + ck0;
          size_t i1 = TRANS_A ? (size_t)ck1 * lda + cm : (size_t)cm * lda + ck1;
          float f0 = Af[i0], f1 = Af[i1];
          if (AMODE == AM_MUL) { f0 *= A2[i0]; f1 *= A2[i1]; }
          else if (AMODE == AM_SUB) { f0 -= A2[i0]; f1 -= A2[i1]; }
          f0 = v0 ? f0 : 0.f;
          f1 = v1 ? f1 : 0.f;
          v = pack_bf16(f0, f1);
        }
        AsU[i] = v;
      }
    }

    // ---------------- stage B tile (f32 -> bf16 pairs, [n][k]) ----------------
    if (interior) {
      for (int i = tid; i < BN * 16; i += 256) {
        int n = i & 255, kp = i >> 8;                 // n fastest -> coalesced loads
        int gk = k0 + 2 * kp;
        float f0 = Bm[(size_t)gk * ldb + n0 + n];
        float f1 = Bm[(size_t)(gk + 1) * ldb + n0 + n];
        BsU[n * 16 + kp] = pack_bf16(f0, f1);
      }
    } else {
      for (int i = tid; i < BN * 16; i += 256) {
        int n = i & 255, kp = i >> 8;
        int gn = n0 + n, gk = k0 + 2 * kp;
        const bool vn = gn < Ncols;
        const bool v0 = vn & (gk < K), v1 = vn & ((gk + 1) < K);
        const int cn  = vn ? gn : Ncols - 1;
        const int ck0 = (gk     < K) ? gk     : K - 1;
        const int ck1 = (gk + 1 < K) ? gk + 1 : K - 1;
        float f0 = Bm[(size_t)ck0 * ldb + cn];
        float f1 = Bm[(size_t)ck1 * ldb + cn];
        f0 = v0 ? f0 : 0.f;
        f1 = v1 ? f1 : 0.f;
        BsU[n * 16 + kp] = pack_bf16(f0, f1);
      }
    }
    __syncthreads();

    // ---------------- compute: 2x4 WMMA tiles per wave ----------------
    #pragma unroll
    for (int rb = 0; rb < 2; ++rb) {
      // A frag: row = l16; lanes0-15 K={0..7,16..23}, lanes16-31 K={8..15,24..31}
      union { v16bf v; unsigned u[8]; } fa;
      const int arow = wm + rb * 16 + l16;
      #pragma unroll
      for (int j = 0; j < 8; ++j)
        fa.u[j] = AsU[arow * 16 + j + (j >= 4 ? 4 : 0) + half * 4];
      #pragma unroll
      for (int cb = 0; cb < 4; ++cb) {
        // B frag: N = l16; lanes0-15 K={0..7,16..23}, lanes16-31 K={8..15,24..31}
        union { v16bf v; unsigned u[8]; } fb;
        const int bn = wn + cb * 16 + l16;
        #pragma unroll
        for (int j = 0; j < 8; ++j)
          fb.u[j] = BsU[bn * 16 + j + (j >= 4 ? 4 : 0) + half * 4];

        acc[rb][cb] = __builtin_amdgcn_wmma_f32_16x16x32_bf16(
            false, fa.v, false, fb.v, (short)0, acc[rb][cb], false, false);
      }
    }
    __syncthreads();
  }

  // ---------------- store: VGPR r -> M = r + 8*half, N = l16 ----------------
  #pragma unroll
  for (int rb = 0; rb < 2; ++rb)
    #pragma unroll
    for (int cb = 0; cb < 4; ++cb)
      #pragma unroll
      for (int r = 0; r < 8; ++r) {
        int gm = m0 + wm + rb * 16 + r + half * 8;
        int gn = n0 + wn + cb * 16 + l16;
        if (gm < M && gn < Ncols) {
          size_t idx = (size_t)gm * ldc + gn;
          float v = acc[rb][cb][r];
          if (ACCUM) v += C[idx];
          if (EPI != EPI_NONE) v += bias[gn];
          if (EPI == EPI_BIAS_RELU) v = fmaxf(v, 0.f);
          C[idx] = v;
        }
      }
}

// ---------------------------------------------------------------------------
// Support kernels
// ---------------------------------------------------------------------------
// one-time f32 -> bf16 pair conversion (adjacency; binary values are exact)
__global__ void cvt_bf16_kernel(unsigned* __restrict__ dst, const float* __restrict__ src,
                                size_t npairs) {
  for (size_t i = (size_t)blockIdx.x * blockDim.x + threadIdx.x; i < npairs;
       i += (size_t)gridDim.x * blockDim.x) {
    float2 f = ((const float2*)src)[i];
    dst[i] = pack_bf16(f.x, f.y);
  }
}

// reciprocal degree norms (hop-invariant): nbw = 1/(1+rowsum(e2n)), nfw = 1/(1+colsum(n2e))
__global__ void norms_kernel(const float* __restrict__ e2n, const float* __restrict__ n2e,
                             float* __restrict__ nbw, float* __restrict__ nfw) {
  int idx = blockIdx.x * blockDim.x + threadIdx.x;
  if (idx >= B_ * N_) return;
  int b = idx / N_, n = idx - b * N_;
  const float* row = e2n + ((size_t)b * N_ + n) * E_;
  float s = 0.f;
  for (int e = 0; e < E_; ++e) s += row[e];
  nbw[idx] = 1.f / (s + 1.f);
  const float* col = n2e + (size_t)b * E_ * N_ + n;
  float s2 = 0.f;
  for (int e = 0; e < E_; ++e) s2 += col[(size_t)e * N_];
  nfw[idx] = 1.f / (s2 + 1.f);
}

// dst = (raw + h) * rnorm[b,n]
__global__ void ew_norm_kernel(float* __restrict__ dst, const float* __restrict__ raw,
                               const float* __restrict__ h, const float* __restrict__ rnorm,
                               size_t n) {
  for (size_t i = (size_t)blockIdx.x * blockDim.x + threadIdx.x; i < n;
       i += (size_t)gridDim.x * blockDim.x)
    dst[i] = (raw[i] + h[i]) * rnorm[i / H_];
}

// agg = (1-sigmoid(z))*fw + sigmoid(z)*bw
__global__ void ew_gated_kernel(float* __restrict__ agg, const float* __restrict__ z,
                                const float* __restrict__ fw, const float* __restrict__ bw,
                                size_t n) {
  for (size_t i = (size_t)blockIdx.x * blockDim.x + threadIdx.x; i < n;
       i += (size_t)gridDim.x * blockDim.x) {
    float s = sigm(z[i]);
    agg[i] = (1.f - s) * fw[i] + s * bw[i];
  }
}

__global__ void ew_sigmoid_kernel(float* __restrict__ x, size_t n) {
  for (size_t i = (size_t)blockIdx.x * blockDim.x + threadIdx.x; i < n;
       i += (size_t)gridDim.x * blockDim.x)
    x[i] = sigm(x[i]);
}

// h = (1-sigmoid(zg))*h + sigmoid(zg)*tanh(tg)
__global__ void ew_gru_kernel(float* __restrict__ h, const float* __restrict__ zg,
                              const float* __restrict__ tg, size_t n) {
  for (size_t i = (size_t)blockIdx.x * blockDim.x + threadIdx.x; i < n;
       i += (size_t)gridDim.x * blockDim.x) {
    float s = sigm(zg[i]);
    h[i] = (1.f - s) * h[i] + s * tanhf(tg[i]);
  }
}

// out2[b,hc] = max_n t[b,n,hc]
__global__ void max_reduce_kernel(float* __restrict__ out2, const float* __restrict__ t) {
  int idx = blockIdx.x * blockDim.x + threadIdx.x;
  if (idx >= B_ * H_) return;
  int b = idx / H_, hc = idx - b * H_;
  const float* p = t + (size_t)b * N_ * H_ + hc;
  float v = p[0];
  for (int n = 1; n < N_; ++n) v = fmaxf(v, p[(size_t)n * H_]);
  out2[idx] = v;
}

// out[n,b,hc] = h[b,n,hc]
__global__ void transpose_out_kernel(float* __restrict__ out, const float* __restrict__ h) {
  size_t total = (size_t)N_ * B_ * H_;
  for (size_t i = (size_t)blockIdx.x * blockDim.x + threadIdx.x; i < total;
       i += (size_t)gridDim.x * blockDim.x) {
    size_t n = i / (B_ * H_);
    size_t r = i - n * (B_ * H_);
    size_t b = r / H_, hc = r - b * H_;
    out[i] = h[((size_t)b * N_ + n) * H_ + hc];
  }
}

// ---------------------------------------------------------------------------
extern "C" void kernel_launch(void* const* d_in, const int* in_sizes, int n_in,
                              void* d_out, int out_size, void* d_ws, size_t ws_size,
                              hipStream_t stream) {
  (void)in_sizes; (void)n_in; (void)out_size; (void)ws_size;

  const float* node_state = (const float*)d_in[0];
  const float* edge_vec   = (const float*)d_in[1];
  const float* n2e        = (const float*)d_in[2];   // [B,E,N]
  const float* e2n        = (const float*)d_in[3];   // [B,N,E]
  const float* W_fuse     = (const float*)d_in[4];   // [2H,H]
  const float* b_fuse     = (const float*)d_in[5];
  const float* W_z        = (const float*)d_in[6];
  const float* W_r        = (const float*)d_in[7];
  const float* W_t        = (const float*)d_in[8];
  const float* W_g        = (const float*)d_in[9];   // [4H,H]
  const float* b_g        = (const float*)d_in[10];
  const float* W_max      = (const float*)d_in[11];  // [H,H]
  float* out = (float*)d_out;

  const size_t NH  = (size_t)B_ * N_ * H_;
  const size_t EH  = (size_t)B_ * E_ * H_;
  const size_t ADJ = (size_t)B_ * E_ * N_;   // elements per adjacency tensor
  const size_t sN  = (size_t)N_ * H_;
  const size_t sE  = (size_t)E_ * H_;
  const size_t sAdj = (size_t)E_ * N_;

  float* ws  = (float*)d_ws;
  float* h   = ws;            // [B,N,H] working node state
  float* abw = h   + NH;
  float* afw = abw + NH;
  float* agg = afw + NH;
  float* zg  = agg + NH;
  float* rg  = zg  + NH;
  float* tg  = rg  + NH;      // also the "raw" temp
  float* e0  = tg  + NH;      // [B,E,H]
  float* e1  = e0  + EH;      // [B,E,H]
  float* nbw = e1  + EH;      // [B,N] reciprocal norms
  float* nfw = nbw + (size_t)B_ * N_;
  unsigned short* n2e_bf = (unsigned short*)(nfw + (size_t)B_ * N_);  // [B,E,N] bf16
  unsigned short* e2n_bf = n2e_bf + ADJ;                              // [B,N,E] bf16

  const float* Wf0 = W_fuse;   const float* Wf1 = W_fuse + (size_t)H_ * H_;
  const float* Wz0 = W_z;      const float* Wz1 = W_z + (size_t)H_ * H_;
  const float* Wr0 = W_r;      const float* Wr1 = W_r + (size_t)H_ * H_;
  const float* Wt0 = W_t;      const float* Wt1 = W_t + (size_t)H_ * H_;
  const float* Wg0 = W_g;
  const float* Wg1 = W_g + 1 * (size_t)H_ * H_;
  const float* Wg2 = W_g + 2 * (size_t)H_ * H_;
  const float* Wg3 = W_g + 3 * (size_t)H_ * H_;

  hipMemcpyAsync(h, node_state, NH * sizeof(float), hipMemcpyDeviceToDevice, stream);

  // one-time: adjacency -> bf16 (exact for binary), degree norms
  cvt_bf16_kernel<<<4096, 256, 0, stream>>>((unsigned*)n2e_bf, n2e, ADJ / 2);
  cvt_bf16_kernel<<<4096, 256, 0, stream>>>((unsigned*)e2n_bf, e2n, ADJ / 2);
  norms_kernel<<<(B_ * N_ + 255) / 256, 256, 0, stream>>>(e2n, n2e, nbw, nfw);

  const int EWB = 2048;
#define GEMM(AT, TA, AMODE, ACC, EPI, Ap, A2p, lda, sA, Bp, ldb, sB, Cp, sC, biasp, M, Nc, K) \
  gemm_bf16_wmma<AT, TA, AMODE, ACC, EPI>                                                     \
      <<<dim3(((M) + 63) / 64, ((Nc) + 255) / 256, B_), 256, 0, stream>>>(                    \
          Ap, A2p, lda, sA, Bp, ldb, sB, Cp, H_, sC, biasp, M, Nc, K)

  for (int hop = 0; hop < HOPS_; ++hop) {
    // ---------- backward message pass ----------
    // e0 = n2e @ h                                   [E x H], K = N
    GEMM(AS_BF16, false, AM_SINGLE, false, EPI_NONE, n2e_bf, nullptr, N_, sAdj,
         h, H_, sN, e0, sE, nullptr, E_, H_, N_);
    // e1 = relu(e0 @ Wf0 + edge_vec @ Wf1 + b_fuse)  (concat-GEMM as K-split)
    GEMM(AS_F32, false, AM_SINGLE, false, EPI_NONE, e0, nullptr, H_, sE,
         Wf0, H_, 0, e1, sE, nullptr, E_, H_, H_);
    GEMM(AS_F32, false, AM_SINGLE, true, EPI_BIAS_RELU, edge_vec, nullptr, H_, sE,
         Wf1, H_, 0, e1, sE, b_fuse, E_, H_, H_);
    // tg = e2n @ e1                                  [N x H], K = E
    GEMM(AS_BF16, false, AM_SINGLE, false, EPI_NONE, e2n_bf, nullptr, E_, sAdj,
         e1, H_, sE, tg, sN, nullptr, N_, H_, E_);
    ew_norm_kernel<<<EWB, 256, 0, stream>>>(abw, tg, h, nbw, NH);

    // ---------- forward message pass (transposed adjacencies) ----------
    // e0 = e2n^T @ h                                 [E x H], K = N
    GEMM(AS_BF16, true, AM_SINGLE, false, EPI_NONE, e2n_bf, nullptr, E_, sAdj,
         h, H_, sN, e0, sE, nullptr, E_, H_, N_);
    GEMM(AS_F32, false, AM_SINGLE, false, EPI_NONE, e0, nullptr, H_, sE,
         Wf0, H_, 0, e1, sE, nullptr, E_, H_, H_);
    GEMM(AS_F32, false, AM_SINGLE, true, EPI_BIAS_RELU, edge_vec, nullptr, H_, sE,
         Wf1, H_, 0, e1, sE, b_fuse, E_, H_, H_);
    // tg = n2e^T @ e1                                [N x H], K = E
    GEMM(AS_BF16, true, AM_SINGLE, false, EPI_NONE, n2e_bf, nullptr, N_, sAdj,
         e1, H_, sE, tg, sN, nullptr, N_, H_, E_);
    ew_norm_kernel<<<EWB, 256, 0, stream>>>(afw, tg, h, nfw, NH);

    // ---------- gated fusion: z = sigmoid([fw,bw,fw*bw,fw-bw] @ W_g + b_g) ----------
    GEMM(AS_F32, false, AM_SINGLE, false, EPI_NONE, afw, nullptr, H_, sN, Wg0, H_, 0, zg, sN, nullptr, N_, H_, H_);
    GEMM(AS_F32, false, AM_SINGLE, true,  EPI_NONE, abw, nullptr, H_, sN, Wg1, H_, 0, zg, sN, nullptr, N_, H_, H_);
    GEMM(AS_F32, false, AM_MUL,    true,  EPI_NONE, afw, abw,     H_, sN, Wg2, H_, 0, zg, sN, nullptr, N_, H_, H_);
    GEMM(AS_F32, false, AM_SUB,    true,  EPI_BIAS, afw, abw,     H_, sN, Wg3, H_, 0, zg, sN, b_g,     N_, H_, H_);
    ew_gated_kernel<<<EWB, 256, 0, stream>>>(agg, zg, afw, abw, NH);

    // ---------- GRU step ----------
    GEMM(AS_F32, false, AM_SINGLE, false, EPI_NONE, h,   nullptr, H_, sN, Wz0, H_, 0, zg, sN, nullptr, N_, H_, H_);
    GEMM(AS_F32, false, AM_SINGLE, true,  EPI_NONE, agg, nullptr, H_, sN, Wz1, H_, 0, zg, sN, nullptr, N_, H_, H_);
    GEMM(AS_F32, false, AM_SINGLE, false, EPI_NONE, h,   nullptr, H_, sN, Wr0, H_, 0, rg, sN, nullptr, N_, H_, H_);
    GEMM(AS_F32, false, AM_SINGLE, true,  EPI_NONE, agg, nullptr, H_, sN, Wr1, H_, 0, rg, sN, nullptr, N_, H_, H_);
    ew_sigmoid_kernel<<<EWB, 256, 0, stream>>>(rg, NH);
    GEMM(AS_F32, false, AM_MUL,    false, EPI_NONE, rg,  h,       H_, sN, Wt0, H_, 0, tg, sN, nullptr, N_, H_, H_);
    GEMM(AS_F32, false, AM_SINGLE, true,  EPI_NONE, agg, nullptr, H_, sN, Wt1, H_, 0, tg, sN, nullptr, N_, H_, H_);
    ew_gru_kernel<<<EWB, 256, 0, stream>>>(h, zg, tg, NH);
  }

  // ---------- graph maxpool + outputs ----------
  GEMM(AS_F32, false, AM_SINGLE, false, EPI_NONE, h, nullptr, H_, sN,
       W_max, H_, 0, tg, sN, nullptr, N_, H_, H_);
  max_reduce_kernel<<<(B_ * H_ + 255) / 256, 256, 0, stream>>>(out + (size_t)N_ * B_ * H_, tg);
  transpose_out_kernel<<<EWB, 256, 0, stream>>>(out, h);
#undef GEMM
}
